// CrossAttnBlockpp_23871428231290
// MI455X (gfx1250) — compile-verified, matrix operand order checked by reading, not run on hardware
//
#include <hip/hip_runtime.h>

// ---------------------------------------------------------------------------
// CDNA5 (gfx1250, wave32) cross-attention block.
// Pipeline: GN -> pack A matrices (bf16) -> 3x WMMA GEMM (qkv) ->
//           flash attention (WMMA QK^T + online softmax + WMMA PV) ->
//           WMMA GEMM out-proj + bias + residual (f32 out).
// Uses GLOBAL_LOAD_ASYNC_TO_LDS_B128 (ASYNCcnt) for transpose-free LDS
// staging when the toolchain exposes the builtin; falls back to VGPR copies.
// ---------------------------------------------------------------------------

typedef __bf16 bf16_t;
typedef __attribute__((ext_vector_type(16))) __bf16 v16bf;
typedef __attribute__((ext_vector_type(8)))  __bf16 v8bf;
typedef __attribute__((ext_vector_type(8)))  float  v8f;
typedef int v4i_t __attribute__((vector_size(16)));

#define NB  16
#define NC  128
#define NCOND 32
#define NHEADS 4
#define HW  1024          // 32*32
#define KIN 160           // C + COND
#define NQKV 512          // C * HEADS
#define NPIX 16384        // B * HW

// --- async global->LDS staging (gfx1250 GLOBAL_LOAD_ASYNC_TO_LDS_B128) -----
#if defined(__has_builtin)
#if __has_builtin(__builtin_amdgcn_global_load_async_to_lds_b128) && \
    __has_builtin(__builtin_amdgcn_s_wait_asynccnt)
#define USE_ASYNC_LDS 1
#endif
#endif
#ifndef USE_ASYNC_LDS
#define USE_ASYNC_LDS 0
#endif

__device__ __forceinline__ void stage_b128(const bf16_t* gsrc, bf16_t* ldst) {
#if USE_ASYNC_LDS
    __builtin_amdgcn_global_load_async_to_lds_b128(
        (__attribute__((address_space(1))) v4i_t*)gsrc,
        (__attribute__((address_space(3))) v4i_t*)ldst,
        /*imm offset=*/0, /*cpol=*/0);
#else
    *(v8bf*)ldst = *(const v8bf*)gsrc;
#endif
}

__device__ __forceinline__ void stage_wait() {
#if USE_ASYNC_LDS
    __builtin_amdgcn_s_wait_asynccnt(0);
#endif
}

__device__ __forceinline__ v8f wmma_bf16(v16bf a, v16bf b, v8f c) {
    return __builtin_amdgcn_wmma_f32_16x16x32_bf16(
        /*neg_a=*/false, a, /*neg_b=*/false, b,
        /*c_mod=*/(short)0, c, /*reuse_a=*/false, /*reuse_b=*/false);
}

// Build a 16x32 bf16 A/B fragment from row-major storage (rowptr points at the
// 32 contiguous K values of this lane's row).
// Layout per ISA 7.12.2: element j -> K = (j>>3)*16 + half*8 + (j&7).
__device__ __forceinline__ v16bf frag_from_row(const bf16_t* rowptr, int half_sel) {
    v8bf c0 = *(const v8bf*)(rowptr + half_sel * 8);
    v8bf c1 = *(const v8bf*)(rowptr + 16 + half_sel * 8);
    v16bf r;
#pragma unroll
    for (int i = 0; i < 8; ++i) { r[i] = c0[i]; r[i + 8] = c1[i]; }
    return r;
}

// ---------------------------------------------------------------------------
// f32 -> bf16 convert (weights)
// ---------------------------------------------------------------------------
__global__ __launch_bounds__(256) void cvt_bf16_kernel(
        const float* __restrict__ in, bf16_t* __restrict__ out, int n) {
    int i = blockIdx.x * 256 + threadIdx.x;
    if (i < n) out[i] = (bf16_t)in[i];
}

// ---------------------------------------------------------------------------
// GroupNorm (32 groups of 4 channels over 32x32) -> bf16 rows of A matrices.
// One block per (b, group); writes columns [g*4, g*4+4) of both Aq and Ak.
// ---------------------------------------------------------------------------
__global__ __launch_bounds__(256) void gn_kernel(
        const float* __restrict__ x, const float* __restrict__ gscale,
        const float* __restrict__ gbias, bf16_t* __restrict__ Aq,
        bf16_t* __restrict__ Ak) {
    int b = blockIdx.x >> 5;
    int g = blockIdx.x & 31;
    int t = threadIdx.x;

    float vals[16];
    float s = 0.f, ss = 0.f;
#pragma unroll
    for (int i = 0; i < 16; ++i) {
        int e = t + i * 256;                 // 0..4095 over (4 ch x 1024 hw)
        int cl = e >> 10, hw = e & 1023;
        float v = x[(((size_t)b * NC) + g * 4 + cl) * HW + hw];
        vals[i] = v; s += v; ss += v * v;
    }
#pragma unroll
    for (int m = 1; m < 32; m <<= 1) {
        s  += __shfl_xor(s, m);
        ss += __shfl_xor(ss, m);
    }
    __shared__ float shs[8], shss[8];
    __shared__ float mean_s, rstd_s;
    int w = t >> 5, lane = t & 31;
    if (lane == 0) { shs[w] = s; shss[w] = ss; }
    __syncthreads();
    if (t == 0) {
        float S = 0.f, SS = 0.f;
        for (int i = 0; i < 8; ++i) { S += shs[i]; SS += shss[i]; }
        float mean = S * (1.f / 4096.f);
        float var  = SS * (1.f / 4096.f) - mean * mean;
        mean_s = mean;
        rstd_s = rsqrtf(var + 1e-6f);
    }
    __syncthreads();
    float mean = mean_s, rstd = rstd_s;
#pragma unroll
    for (int i = 0; i < 16; ++i) {
        int e = t + i * 256;
        int cl = e >> 10, hw = e & 1023;
        int c = g * 4 + cl;
        float y = (vals[i] - mean) * rstd * gscale[c] + gbias[c];
        bf16_t yb = (bf16_t)y;
        size_t off = ((size_t)b * HW + hw) * KIN + c;
        Aq[off] = yb;
        Ak[off] = yb;
    }
}

// ---------------------------------------------------------------------------
// Pack conditioning channels into A-matrix columns [128,160).
// q: repeat(q_cond,2). k: alternate k_a/k_b per pair member.
// ---------------------------------------------------------------------------
__global__ __launch_bounds__(256) void condpack_kernel(
        const float* __restrict__ qc, const float* __restrict__ ka,
        const float* __restrict__ kb, bf16_t* __restrict__ Aq,
        bf16_t* __restrict__ Ak) {
    size_t gi = (size_t)blockIdx.x * 256 + threadIdx.x;   // < 16*32*1024
    int hw = (int)(gi & 1023);
    int cc = (int)((gi >> 10) & 31);
    int b  = (int)(gi >> 15);
    size_t src = (((size_t)(b >> 1)) * NCOND + cc) * HW + hw;
    size_t dst = ((size_t)b * HW + hw) * KIN + NC + cc;
    Aq[dst] = (bf16_t)qc[src];
    Ak[dst] = (bf16_t)((b & 1) ? kb[src] : ka[src]);
}

// ---------------------------------------------------------------------------
// QKV GEMM: [16384 x 160] (bf16, row-major) x [160 x 512] (bf16, row-major)
// + bias, output scattered to channel-major bf16 [b][n][hw].
// Block tile 128(M) x 64(N), 8 waves (4x2), each wave 32x32 (2x2 WMMA tiles).
// A tile staged with async global->LDS; B tile transposed via VGPRs.
// ---------------------------------------------------------------------------
__global__ __launch_bounds__(256) void gemm_qkv_kernel(
        const bf16_t* __restrict__ A, const bf16_t* __restrict__ Wb,
        const float* __restrict__ bias, bf16_t* __restrict__ outCHW) {
    __shared__ __align__(16) bf16_t lA[128 * 32];   // [m][k]
    __shared__ __align__(16) bf16_t lB[64 * 32];    // [n][k] (transposed)

    int bm = blockIdx.x >> 3;          // 0..127
    int bn = blockIdx.x & 7;           // 0..7
    int mbase = bm * 128, nbase = bn * 64;
    int t = threadIdx.x, w = t >> 5, lane = t & 31;
    int half = lane >> 4, ln = lane & 15;
    int wm = (w & 3) * 32, wn = (w >> 2) * 32;

    v8f acc[2][2] = {};

    for (int ks = 0; ks < KIN; ks += 32) {
        {   // stage A tile: 128 rows x 32 k, 32 bytes per thread (async)
            int row = t >> 1, colh = t & 1;
            const bf16_t* src = A + (size_t)(mbase + row) * KIN + ks + colh * 16;
            bf16_t* dst = lA + row * 32 + colh * 16;
            stage_b128(src, dst);
            stage_b128(src + 8, dst + 8);
        }
        {   // stage B tile transposed: 32 k x 64 n
            int krow = t >> 3, n0 = (t & 7) * 8;
            const bf16_t* src = Wb + (size_t)(ks + krow) * NQKV + nbase + n0;
            v8bf v = *(const v8bf*)src;
#pragma unroll
            for (int i = 0; i < 8; ++i) lB[(n0 + i) * 32 + krow] = v[i];
        }
        stage_wait();
        __syncthreads();

        v16bf af[2], bfr[2];
#pragma unroll
        for (int mt = 0; mt < 2; ++mt)
            af[mt] = frag_from_row(lA + (wm + mt * 16 + ln) * 32, half);
#pragma unroll
        for (int nt = 0; nt < 2; ++nt)
            bfr[nt] = frag_from_row(lB + (wn + nt * 16 + ln) * 32, half);
#pragma unroll
        for (int mt = 0; mt < 2; ++mt)
#pragma unroll
            for (int nt = 0; nt < 2; ++nt)
                acc[mt][nt] = wmma_bf16(af[mt], bfr[nt], acc[mt][nt]);
        __syncthreads();
    }

    int b = mbase >> 10;    // 8 M-blocks per image -> constant per block
#pragma unroll
    for (int mt = 0; mt < 2; ++mt) {
#pragma unroll
        for (int nt = 0; nt < 2; ++nt) {
            int nglob = nbase + wn + nt * 16 + ln;
            float bv = bias[nglob];
#pragma unroll
            for (int r = 0; r < 8; ++r) {
                int m = mbase + wm + mt * 16 + r + 8 * half;
                int hw = m & 1023;
                outCHW[((size_t)b * NQKV + nglob) * HW + hw] =
                    (bf16_t)(acc[mt][nt][r] + bv);
            }
        }
    }
}

// ---------------------------------------------------------------------------
// Flash attention per (pair, member, head, 128-query-row tile).
// Q (128ch) frags preloaded in regs; K/V 32-col chunks staged in LDS shared by
// 8 waves (V via async global->LDS); online softmax with half-wave shfl
// reductions; P re-swizzled through per-wave LDS; output bf16 rows of A3.
// ---------------------------------------------------------------------------
__global__ __launch_bounds__(256) void attn_kernel(
        const bf16_t* __restrict__ Qb, const bf16_t* __restrict__ Kb,
        const bf16_t* __restrict__ Vb, bf16_t* __restrict__ A3) {
    __shared__ __align__(16) bf16_t lK[32 * 128];    // [kv][c]
    __shared__ __align__(16) bf16_t lV[128 * 32];    // [c][kv]
    __shared__ __align__(16) bf16_t lP[8][16 * 32];  // per-wave P staging

    int idx = blockIdx.x;
    int qt = idx & 7;  idx >>= 3;        // query tile (128 rows)
    int e  = idx & 3;  idx >>= 2;        // head
    int mq = idx & 1;                    // pair member for q
    int p  = idx >> 1;                   // pair
    int bq = 2 * p + mq;
    int bk = 2 * p + 1 - mq;             // cross conditioning

    const bf16_t* qptr = Qb + ((size_t)bq * NQKV + e * NC) * HW;
    const bf16_t* kptr = Kb + ((size_t)bk * NQKV + e * NC) * HW;
    const bf16_t* vptr = Vb + ((size_t)bk * NQKV + e * NC) * HW;

    int t = threadIdx.x, w = t >> 5, lane = t & 31;
    int half = lane >> 4, ln = lane & 15;
    const float scale = 0.08838834764831845f;   // 128^-0.5

    // Preload the 4 Q A-fragments (16 q rows x 128 channels), reused 32x.
    int row_hw = qt * 128 + w * 16 + ln;
    v16bf aq[4];
#pragma unroll
    for (int cs = 0; cs < 4; ++cs) {
        v16bf f;
#pragma unroll
        for (int j = 0; j < 16; ++j) {
            int c = cs * 32 + ((j >> 3) << 4) + half * 8 + (j & 7);
            f[j] = qptr[(size_t)c * HW + row_hw];
        }
        aq[cs] = f;
    }

    v8f o[8] = {};
    float mrun[8], lrun[8];
#pragma unroll
    for (int r = 0; r < 8; ++r) { mrun[r] = -1e30f; lrun[r] = 0.f; }

    for (int kb = 0; kb < HW; kb += 32) {
        {   // stage K chunk transposed: [kv][c]
            int c = t >> 1, kv0 = (t & 1) * 16;
            const bf16_t* src = kptr + (size_t)c * HW + kb + kv0;
            v8bf v0 = *(const v8bf*)src, v1 = *(const v8bf*)(src + 8);
#pragma unroll
            for (int i = 0; i < 8; ++i) {
                lK[(kv0 + i) * 128 + c]     = v0[i];
                lK[(kv0 + 8 + i) * 128 + c] = v1[i];
            }
        }
        {   // stage V chunk: [c][kv] (contiguous -> async global->LDS)
            int c = t >> 1, kv0 = (t & 1) * 16;
            const bf16_t* src = vptr + (size_t)c * HW + kb + kv0;
            bf16_t* dst = lV + c * 32 + kv0;
            stage_b128(src, dst);
            stage_b128(src + 8, dst + 8);
        }
        stage_wait();
        __syncthreads();

        // S = Q^T K : 16 x 32 (two 16x16 C tiles), K-dim = 128 channels.
        v8f s0 = {}, s1 = {};
#pragma unroll
        for (int cs = 0; cs < 4; ++cs) {
            v16bf b0 = frag_from_row(lK + (0 * 16 + ln) * 128 + cs * 32, half);
            v16bf b1 = frag_from_row(lK + (1 * 16 + ln) * 128 + cs * 32, half);
            s0 = wmma_bf16(aq[cs], b0, s0);
            s1 = wmma_bf16(aq[cs], b1, s1);
        }
        s0 = s0 * scale;
        s1 = s1 * scale;

        // Online softmax (rows live on 16-lane halves; reduce masks 1..8).
#pragma unroll
        for (int r = 0; r < 8; ++r) {
            float v = fmaxf(s0[r], s1[r]);
            v = fmaxf(v, __shfl_xor(v, 1));
            v = fmaxf(v, __shfl_xor(v, 2));
            v = fmaxf(v, __shfl_xor(v, 4));
            v = fmaxf(v, __shfl_xor(v, 8));
            float mnew  = fmaxf(mrun[r], v);
            float alpha = __expf(mrun[r] - mnew);
            float p0 = __expf(s0[r] - mnew);
            float p1 = __expf(s1[r] - mnew);
            float rs = p0 + p1;
            rs += __shfl_xor(rs, 1);
            rs += __shfl_xor(rs, 2);
            rs += __shfl_xor(rs, 4);
            rs += __shfl_xor(rs, 8);
            lrun[r] = lrun[r] * alpha + rs;
            mrun[r] = mnew;
            s0[r] = p0; s1[r] = p1;
#pragma unroll
            for (int nt = 0; nt < 8; ++nt) o[nt][r] *= alpha;
        }

        // Re-swizzle P: C-layout -> row-major [row][32] in per-wave LDS,
        // then load back as an A fragment (same-wave DS ops are in order).
        {
            bf16_t* pp = lP[w];
#pragma unroll
            for (int r = 0; r < 8; ++r) {
                int row = r + 8 * half;
                pp[row * 32 + ln]      = (bf16_t)s0[r];
                pp[row * 32 + 16 + ln] = (bf16_t)s1[r];
            }
        }
        v16bf pf = frag_from_row(lP[w] + ln * 32, half);

        // O += P * V^T (8 channel tiles).
#pragma unroll
        for (int nt = 0; nt < 8; ++nt) {
            v16bf bv = frag_from_row(lV + (nt * 16 + ln) * 32, half);
            o[nt] = wmma_bf16(pf, bv, o[nt]);
        }
        __syncthreads();
    }

    // Normalize and emit rows of A3 [pixel][512] (head-major channels).
#pragma unroll
    for (int r = 0; r < 8; ++r) {
        float inv = 1.f / lrun[r];
        int hw = qt * 128 + w * 16 + r + 8 * half;
        size_t rowbase = ((size_t)bq * HW + hw) * NQKV + e * NC;
#pragma unroll
        for (int nt = 0; nt < 8; ++nt)
            A3[rowbase + nt * 16 + ln] = (bf16_t)(o[nt][r] * inv);
    }
}

// ---------------------------------------------------------------------------
// Output projection: [16384 x 512] x [512 x 128] + b3 + residual x -> f32.
// Block tile 128(M) x 128(N); each wave: 16 rows x 128 cols (8 WMMA tiles).
// ---------------------------------------------------------------------------
__global__ __launch_bounds__(256) void gemm_final_kernel(
        const bf16_t* __restrict__ A3, const bf16_t* __restrict__ W3b,
        const float* __restrict__ b3, const float* __restrict__ x,
        float* __restrict__ out) {
    __shared__ __align__(16) bf16_t lB[128 * 32];   // [n][k] transposed

    int mbase = blockIdx.x * 128;
    int t = threadIdx.x, w = t >> 5, lane = t & 31;
    int half = lane >> 4, ln = lane & 15;
    int mrow = mbase + w * 16 + ln;

    v8f acc[8] = {};

    for (int ks = 0; ks < NQKV; ks += 32) {
        {   // stage W3 tile transposed: 32 k x 128 n
            int krow = t >> 3, n0 = (t & 7) * 16;
            const bf16_t* src = W3b + (size_t)(ks + krow) * NC + n0;
            v8bf v0 = *(const v8bf*)src, v1 = *(const v8bf*)(src + 8);
#pragma unroll
            for (int i = 0; i < 8; ++i) {
                lB[(n0 + i) * 32 + krow]     = v0[i];
                lB[(n0 + 8 + i) * 32 + krow] = v1[i];
            }
        }
        __syncthreads();

        v16bf af = frag_from_row(A3 + (size_t)mrow * NQKV + ks, half);
#pragma unroll
        for (int nt = 0; nt < 8; ++nt) {
            v16bf bfr = frag_from_row(lB + (nt * 16 + ln) * 32, half);
            acc[nt] = wmma_bf16(af, bfr, acc[nt]);
        }
        __syncthreads();
    }

    int b = mbase >> 10;
#pragma unroll
    for (int nt = 0; nt < 8; ++nt) {
        int c = nt * 16 + ln;
        float bv = b3[c];
#pragma unroll
        for (int r = 0; r < 8; ++r) {
            int m = mbase + w * 16 + r + 8 * half;
            int hw = m & 1023;
            size_t oi = ((size_t)b * NC + c) * HW + hw;
            out[oi] = x[oi] + acc[nt][r] + bv;
        }
    }
}

// ---------------------------------------------------------------------------
// Host-side launch
// ---------------------------------------------------------------------------
extern "C" void kernel_launch(void* const* d_in, const int* in_sizes, int n_in,
                              void* d_out, int out_size, void* d_ws, size_t ws_size,
                              hipStream_t stream) {
    const float* x  = (const float*)d_in[0];
    const float* qc = (const float*)d_in[1];
    const float* ka = (const float*)d_in[2];
    const float* kb = (const float*)d_in[3];
    const float* gs = (const float*)d_in[4];
    const float* gb = (const float*)d_in[5];
    const float* W0 = (const float*)d_in[6];
    const float* b0 = (const float*)d_in[7];
    const float* W1 = (const float*)d_in[8];
    const float* b1 = (const float*)d_in[9];
    const float* W2 = (const float*)d_in[10];
    const float* b2 = (const float*)d_in[11];
    const float* W3 = (const float*)d_in[12];
    const float* b3 = (const float*)d_in[13];
    float* out = (float*)d_out;

    char* ws = (char*)d_ws;
    size_t off = 0;
    auto alloc = [&](size_t bytes) -> char* {
        char* ptr = ws + off;
        off = (off + bytes + 255) & ~(size_t)255;
        return ptr;
    };
    bf16_t* Aq  = (bf16_t*)alloc((size_t)NPIX * KIN * 2);
    bf16_t* Ak  = (bf16_t*)alloc((size_t)NPIX * KIN * 2);
    bf16_t* W0b = (bf16_t*)alloc((size_t)KIN * NQKV * 2);
    bf16_t* W1b = (bf16_t*)alloc((size_t)KIN * NQKV * 2);
    bf16_t* W2b = (bf16_t*)alloc((size_t)KIN * NQKV * 2);
    bf16_t* W3b = (bf16_t*)alloc((size_t)NQKV * NC * 2);
    bf16_t* Qb  = (bf16_t*)alloc((size_t)NB * NQKV * HW * 2);
    bf16_t* Kbm = (bf16_t*)alloc((size_t)NB * NQKV * HW * 2);
    bf16_t* Vb  = (bf16_t*)alloc((size_t)NB * NQKV * HW * 2);
    bf16_t* A3  = (bf16_t*)alloc((size_t)NPIX * NQKV * 2);
    (void)ws_size; (void)in_sizes; (void)n_in; (void)out_size;

    // Weight conversion
    cvt_bf16_kernel<<<(KIN * NQKV + 255) / 256, 256, 0, stream>>>(W0, W0b, KIN * NQKV);
    cvt_bf16_kernel<<<(KIN * NQKV + 255) / 256, 256, 0, stream>>>(W1, W1b, KIN * NQKV);
    cvt_bf16_kernel<<<(KIN * NQKV + 255) / 256, 256, 0, stream>>>(W2, W2b, KIN * NQKV);
    cvt_bf16_kernel<<<(NQKV * NC + 255) / 256, 256, 0, stream>>>(W3, W3b, NQKV * NC);

    // GroupNorm + A-matrix packing
    gn_kernel<<<NB * 32, 256, 0, stream>>>(x, gs, gb, Aq, Ak);
    condpack_kernel<<<(NB * NCOND * HW) / 256, 256, 0, stream>>>(qc, ka, kb, Aq, Ak);

    // QKV projections
    gemm_qkv_kernel<<<(NPIX / 128) * (NQKV / 64), 256, 0, stream>>>(Aq, W0b, b0, Qb);
    gemm_qkv_kernel<<<(NPIX / 128) * (NQKV / 64), 256, 0, stream>>>(Ak, W1b, b1, Kbm);
    gemm_qkv_kernel<<<(NPIX / 128) * (NQKV / 64), 256, 0, stream>>>(Ak, W2b, b2, Vb);

    // Cross attention: 8 pairs x 2 directions x 4 heads x 8 query tiles
    attn_kernel<<<8 * 2 * 4 * 8, 256, 0, stream>>>(Qb, Kbm, Vb, A3);

    // Output projection + residual
    gemm_final_kernel<<<NPIX / 128, 256, 0, stream>>>(A3, W3b, b3, x, out);
}
